// SeparationLoss_5007931867181
// MI455X (gfx1250) — matrix operand order repaired and checked by reading, not compiled, not run on hardware
//
#include <hip/hip_runtime.h>

#define B_ 16
#define L_ 64
#define K_ 32
#define C_ 32
#define D_ 512
#define KC_ (K_ * C_)
#define NPAIR_ ((L_ * (L_ - 1)) / 2)   // 2016 pairs (i<j) per batch

typedef __attribute__((ext_vector_type(16))) __bf16 v16bf;
typedef __attribute__((ext_vector_type(8)))  float  v8f;

// Native converts: fptrunc f32->bf16 is RNE and lowers to v_cvt_pk_bf16_f32
// on gfx1250 (one VALU op per packed pair) instead of a bit-twiddle sequence.
__device__ __forceinline__ __bf16 f32_to_bf16(float x) {
  return static_cast<__bf16>(x);
}
__device__ __forceinline__ float bf16_to_f32(__bf16 b) {
  return static_cast<float>(b);
}

// ---- zero the 4 global accumulators ----
__global__ void k_init(float* __restrict__ acc) {
  if (threadIdx.x < 4) acc[threadIdx.x] = 0.0f;
}

// ---- log_softmax over C=32: one wave32 per row, 8 rows per 256-thread block ----
__global__ void k_logsoftmax(const float* __restrict__ prior,
                             float* __restrict__ logp, float* __restrict__ p) {
  int lane = threadIdx.x & 31;
  int row  = blockIdx.x * 8 + (threadIdx.x >> 5);   // row < B*L*K = 32768
  int idx  = row * C_ + lane;
  float x = prior[idx];
  float m = x;
  #pragma unroll
  for (int off = 16; off; off >>= 1) m = fmaxf(m, __shfl_xor(m, off, 32));
  float e = __expf(x - m);
  float s = e;
  #pragma unroll
  for (int off = 16; off; off >>= 1) s += __shfl_xor(s, off, 32);
  float lse = m + __logf(s);
  float lp  = x - lse;
  logp[idx] = lp;
  p[idx]    = __expf(lp);
}

// ---- squared norms of h rows: one wave32 per row of D=512 ----
__global__ void k_sqnorm(const float* __restrict__ h, float* __restrict__ sqn) {
  int lane = threadIdx.x & 31;
  int row  = blockIdx.x * 8 + (threadIdx.x >> 5);   // row < B*L = 1024
  const float* hr = h + (size_t)row * D_;
  float s = 0.0f;
  #pragma unroll
  for (int i = 0; i < D_ / 32; ++i) { float v = hr[lane + i * 32]; s += v * v; }
  #pragma unroll
  for (int off = 16; off; off >>= 1) s += __shfl_xor(s, off, 32);
  if (lane == 0) sqn[row] = s;
}

// ---- per-batch Gram matrix H * H^T via split-bf16 WMMA ----
// grid = B*16 blocks of one wave32; each wave owns one 16x16 output tile.
__global__ void __launch_bounds__(32)
k_gram(const float* __restrict__ h, float* __restrict__ gram) {
  int b    = blockIdx.x >> 4;
  int tile = blockIdx.x & 15;
  int it = tile >> 2, jt = tile & 3;
  int lane = threadIdx.x;
  int half = lane >> 4;       // 0: lanes 0-15, 1: lanes 16-31
  int l15  = lane & 15;

  const float* Hb   = h + (size_t)b * L_ * D_;
  const float* arow = Hb + (size_t)(it * 16 + l15) * D_;  // A: M = it*16 + l15
  const float* brow = Hb + (size_t)(jt * 16 + l15) * D_;  // B: N = jt*16 + l15

  v8f acc = {};
  for (int k0 = 0; k0 < D_; k0 += 32) {
    // A-matrix 16x32 bf16 lane layout: lane<16 -> K = {0..7, 16..23}+k0,
    //                                  lane>=16 -> K = {8..15, 24..31}+k0
    float af[16], bf[16];
    const float* ap = arow + k0 + half * 8;
    #pragma unroll
    for (int e = 0; e < 8; ++e) { af[e] = ap[e]; af[8 + e] = ap[16 + e]; }
    // B-matrix 32x16 bf16 lane layout: lane<16 -> K = {0..7, 16..23}+k0,
    //                                  lane>=16 -> K = {8..15, 24..31}+k0 (N = l15)
    const float* bp = brow + k0 + half * 8;
    #pragma unroll
    for (int e = 0; e < 8; ++e) { bf[e] = bp[e]; bf[8 + e] = bp[16 + e]; }

    v16bf ahi, alo, bhi, blo;
    #pragma unroll
    for (int e = 0; e < 16; ++e) {
      __bf16 h1 = f32_to_bf16(af[e]);
      ahi[e] = h1;
      alo[e] = f32_to_bf16(af[e] - bf16_to_f32(h1));
      __bf16 h2 = f32_to_bf16(bf[e]);
      bhi[e] = h2;
      blo[e] = f32_to_bf16(bf[e] - bf16_to_f32(h2));
    }
    // h_i . h_j ~= hi.hi + hi.lo + lo.hi   (lo.lo dropped, ~2^-16 rel error)
    acc = __builtin_amdgcn_wmma_f32_16x16x32_bf16(false, ahi, false, bhi,
                                                  (short)0, acc, false, false);
    acc = __builtin_amdgcn_wmma_f32_16x16x32_bf16(false, ahi, false, blo,
                                                  (short)0, acc, false, false);
    acc = __builtin_amdgcn_wmma_f32_16x16x32_bf16(false, alo, false, bhi,
                                                  (short)0, acc, false, false);
  }
  // D layout: VGPR r -> M = r + 8*half ; N = lane&15
  #pragma unroll
  for (int r = 0; r < 8; ++r) {
    int M = it * 16 + half * 8 + r;
    int N = jt * 16 + l15;
    gram[((size_t)b * L_ + M) * L_ + N] = acc[r];
  }
}

// ---- per-pair JSD + hard masks + accumulation (one 256-thread block per pair) ----
__global__ void k_pair(const float* __restrict__ logp, const float* __restrict__ p,
                       const float* __restrict__ gram, const float* __restrict__ sqn,
                       float* __restrict__ acc) {
  int blk = blockIdx.x;
  int b = blk / NPAIR_;
  int t = blk % NPAIR_;
  // decode upper-triangular pair (i < j)
  int i = 0, cnt = L_ - 1;
  while (t >= cnt) { t -= cnt; ++i; --cnt; }
  int j = i + 1 + t;

  size_t base1 = ((size_t)b * L_ + i) * KC_;
  size_t base2 = ((size_t)b * L_ + j) * KC_;
  float local = 0.0f;
  for (int u = threadIdx.x; u < KC_; u += 256) {
    float p1  = p[base1 + u],    p2  = p[base2 + u];
    float lp1 = logp[base1 + u], lp2 = logp[base2 + u];
    float logm = __logf(0.5f * (p1 + p2));
    local += 0.5f * (p1 * (lp1 - logm) + p2 * (lp2 - logm));
  }
  #pragma unroll
  for (int off = 16; off; off >>= 1) local += __shfl_xor(local, off, 32);
  __shared__ float sred[8];
  if ((threadIdx.x & 31) == 0) sred[threadIdx.x >> 5] = local;
  __syncthreads();
  if (threadIdx.x == 0) {
    float jsd = 0.0f;
    #pragma unroll
    for (int w = 0; w < 8; ++w) jsd += sred[w];   // SCALING_FACTOR = 1
    float g   = gram[((size_t)b * L_ + i) * L_ + j];
    float mse = (sqn[b * L_ + i] + sqn[b * L_ + j] - 2.0f * g) * (1.0f / D_);
    if (jsd < 5.0f) {                 // hard attract mask (STE forward == hard)
      atomicAdd(&acc[0], mse);
      atomicAdd(&acc[1], 1.0f);
    } else {                          // hard repulse mask
      atomicAdd(&acc[2], __expf(-mse));   // DIST_TEMPERATURE = 1
      atomicAdd(&acc[3], 1.0f);
    }
  }
}

// ---- finalize the two scalar losses ----
__global__ void k_final(const float* __restrict__ acc, float* __restrict__ out) {
  if (threadIdx.x == 0) {
    float la = acc[0] / (acc[1] + 1e-10f);
    out[0] = (la > 0.0f) ? la : 0.0f;   // ATT_LOSS_GATE = 0
    float lr = acc[2] / (acc[3] + 1e-10f);
    out[1] = (lr > 0.0f) ? lr : 0.0f;   // REP_LOSS_GATE = 0
  }
}

extern "C" void kernel_launch(void* const* d_in, const int* in_sizes, int n_in,
                              void* d_out, int out_size, void* d_ws, size_t ws_size,
                              hipStream_t stream) {
  (void)in_sizes; (void)n_in; (void)out_size; (void)ws_size;
  const float* prior = (const float*)d_in[0];   // (16, 64, 32, 32) f32
  const float* h     = (const float*)d_in[1];   // (16, 64, 512)    f32
  float* out = (float*)d_out;                   // (loss_att, loss_rep)

  float* ws   = (float*)d_ws;
  float* logp = ws;                                   // 1,048,576 f32
  float* p    = logp + (size_t)B_ * L_ * K_ * C_;     // 1,048,576 f32
  float* gram = p    + (size_t)B_ * L_ * K_ * C_;     //    65,536 f32
  float* sqn  = gram + (size_t)B_ * L_ * L_;          //     1,024 f32
  float* acc  = sqn  + (size_t)B_ * L_;               //         4 f32

  k_init<<<1, 32, 0, stream>>>(acc);
  k_logsoftmax<<<(B_ * L_ * K_) / 8, 256, 0, stream>>>(prior, logp, p);
  k_sqnorm<<<(B_ * L_) / 8, 256, 0, stream>>>(h, sqn);
  k_gram<<<B_ * 16, 32, 0, stream>>>(h, gram);
  k_pair<<<B_ * NPAIR_, 256, 0, stream>>>(logp, p, gram, sqn, acc);
  k_final<<<1, 32, 0, stream>>>(acc, out);
}
